// VectorQuantizerEMA_42949672960042
// MI455X (gfx1250) — matrix operand order repaired and checked by reading, not compiled, not run on hardware
//
#include <hip/hip_runtime.h>
#include <cstdint>

typedef __attribute__((ext_vector_type(16))) __bf16 v16bf;
typedef __attribute__((ext_vector_type(8)))  float  v8f;

#define N_TOKENS 65536
#define NUM_EMB  1024
#define EMB_DIM  128

// -------- workspace layout (in floats) --------
// [0, 1024)            eSumSq    f32
// [1024, 2048)         counts    u32
// [2048, 10240)        partials  f32 (8192 blocks)
// [10240, 75776)       idx       i32 (65536)
// [75776, 141312)      packedBhi u32 (65536)  -- bf16-pair codebook, WMMA B layout
// [141312, 206848)     packedBlo u32 (65536)

__global__ void k_zero2(float2* __restrict__ p, long long n2) {
    long long i = (long long)blockIdx.x * blockDim.x + threadIdx.x;
    long long stride = (long long)gridDim.x * blockDim.x;
    float2 z; z.x = 0.0f; z.y = 0.0f;
    for (; i < n2; i += stride) p[i] = z;
}

__global__ void k_esumsq(const float* __restrict__ emb, float* __restrict__ esq) {
    int n = blockIdx.x * blockDim.x + threadIdx.x;
    if (n < NUM_EMB) {
        const float* row = emb + (size_t)n * EMB_DIM;
        float s = 0.0f;
        #pragma unroll 4
        for (int d = 0; d < EMB_DIM; ++d) s += row[d] * row[d];
        esq[n] = s;
    }
}

// Pack codebook into bf16 hi/lo pairs, pre-swizzled into per-lane WMMA B-operand order:
// u32 index = ((nc*4 + c)*32 + lane)*8 + r
// lane = (h<<4)|m : column n = nc*16+m, K pair = c*32 + h*16 + 2r (pair packed lo|hi<<16)
__global__ void k_pack(const float* __restrict__ emb,
                       unsigned* __restrict__ bhi, unsigned* __restrict__ blo) {
    const int gid = blockIdx.x * blockDim.x + threadIdx.x;   // 0..65535
    const int r    = gid & 7;
    const int lane = (gid >> 3) & 31;
    const int c    = (gid >> 8) & 3;
    const int nc   = gid >> 10;
    const int m = lane & 15, h = lane >> 4;
    const int n  = nc * 16 + m;
    const int kk = c * 32 + h * 16 + 2 * r;
    const float f0 = emb[n * EMB_DIM + kk];
    const float f1 = emb[n * EMB_DIM + kk + 1];
    const __bf16 h0 = (__bf16)f0, h1 = (__bf16)f1;
    const __bf16 l0 = (__bf16)(f0 - (float)h0), l1 = (__bf16)(f1 - (float)h1);
    const unsigned uh0 = (unsigned)__builtin_bit_cast(unsigned short, h0);
    const unsigned uh1 = (unsigned)__builtin_bit_cast(unsigned short, h1);
    const unsigned ul0 = (unsigned)__builtin_bit_cast(unsigned short, l0);
    const unsigned ul1 = (unsigned)__builtin_bit_cast(unsigned short, l1);
    bhi[gid] = (uh1 << 16) | uh0;
    blo[gid] = (ul1 << 16) | ul0;
}

// 8 waves/block; each wave argmins over all 1024 codes for 16 rows.
// Distance surrogate: ||e||^2 - 2 x.e ; x.e via bf16 hi/lo split (xh*eh + xh*el + xl*eh).
// B operands come pre-packed from ws: hot loop = b128 loads + WMMA only.
__global__ void __launch_bounds__(256) k_argmin(const float* __restrict__ x,
                                                const v16bf* __restrict__ bhiV,
                                                const v16bf* __restrict__ bloV,
                                                const float* __restrict__ esq,
                                                int* __restrict__ idxOut) {
    const int lane = threadIdx.x & 31;
    const int wave = threadIdx.x >> 5;
    const int m = lane & 15;
    const int h = lane >> 4;
    const int rowBase = blockIdx.x * 128 + wave * 16;

    // ---- convert this wave's A tile once (amortized over 64 chunks) ----
    v16bf a_hi[4], a_lo[4];
    const float* arow = x + (size_t)(rowBase + m) * EMB_DIM;
    #pragma unroll
    for (int c = 0; c < 4; ++c) {
        #pragma unroll
        for (int r = 0; r < 8; ++r) {
            const int kk = c * 32 + h * 16 + 2 * r;
            float f0 = arow[kk], f1 = arow[kk + 1];
            __bf16 b0 = (__bf16)f0, b1 = (__bf16)f1;
            a_hi[c][2 * r]     = b0;
            a_hi[c][2 * r + 1] = b1;
            a_lo[c][2 * r]     = (__bf16)(f0 - (float)b0);
            a_lo[c][2 * r + 1] = (__bf16)(f1 - (float)b1);
        }
    }

    float bestD[8];
    int   bestI[8];
    #pragma unroll
    for (int j = 0; j < 8; ++j) { bestD[j] = 3.4e38f; bestI[j] = 0; }

    for (int nc = 0; nc < NUM_EMB / 16; ++nc) {
        const int nIdx = nc * 16 + m;
        // two independent accumulator chains (6 WMMAs each) for XDL ILP
        v8f acc0 = {0.f, 0.f, 0.f, 0.f, 0.f, 0.f, 0.f, 0.f};
        v8f acc1 = {0.f, 0.f, 0.f, 0.f, 0.f, 0.f, 0.f, 0.f};
        #pragma unroll
        for (int c = 0; c < 4; ++c) {
            const int bi = ((nc * 4 + c) << 5) + lane;  // 32-byte vector index
            v16bf b_hi = bhiV[bi];
            v16bf b_lo = bloV[bi];
            acc0 = __builtin_amdgcn_wmma_f32_16x16x32_bf16(false, a_hi[c], false, b_hi,
                                                           (short)0, acc0, false, false);
            acc1 = __builtin_amdgcn_wmma_f32_16x16x32_bf16(false, a_lo[c], false, b_hi,
                                                           (short)0, acc1, false, false);
            if (c < 2)
                acc0 = __builtin_amdgcn_wmma_f32_16x16x32_bf16(false, a_hi[c], false, b_lo,
                                                               (short)0, acc0, false, false);
            else
                acc1 = __builtin_amdgcn_wmma_f32_16x16x32_bf16(false, a_hi[c], false, b_lo,
                                                               (short)0, acc1, false, false);
        }
        const float es = esq[nIdx];
        // C layout: lane half h, VGPR j -> row h*8+j, column = m (this lane's nIdx)
        #pragma unroll
        for (int j = 0; j < 8; ++j) {
            float d = es - 2.0f * (acc0[j] + acc1[j]);
            if (d < bestD[j]) { bestD[j] = d; bestI[j] = nIdx; }  // '<' keeps smallest index
        }
    }

    // argmin across the 16 lanes of each half (wave32 shuffles, masks <= 8 stay in half)
    #pragma unroll
    for (int j = 0; j < 8; ++j) {
        #pragma unroll
        for (int off = 8; off >= 1; off >>= 1) {
            float od = __shfl_xor(bestD[j], off, 32);
            int   oi = __shfl_xor(bestI[j], off, 32);
            if (od < bestD[j] || (od == bestD[j] && oi < bestI[j])) {
                bestD[j] = od; bestI[j] = oi;
            }
        }
    }
    if (m == 0) {
        #pragma unroll
        for (int j = 0; j < 8; ++j)
            idxOut[rowBase + h * 8 + j] = bestI[j];
    }
}

// one wave per row: gather codebook row, write quantized_st, deterministic loss partials,
// histogram atomics, one-hot scatter.
__global__ void __launch_bounds__(256) k_finalize(const float* __restrict__ x,
                                                  const float* __restrict__ emb,
                                                  const int* __restrict__ idxArr,
                                                  float* __restrict__ quant,
                                                  float* __restrict__ enc,
                                                  unsigned* __restrict__ counts,
                                                  float* __restrict__ partials) {
    __shared__ float wsum[8];
    const int lane = threadIdx.x & 31;
    const int wave = threadIdx.x >> 5;
    const int row = blockIdx.x * 8 + wave;
    const int idx = idxArr[row];
    const float* er = emb + (size_t)idx * EMB_DIM;
    const float* xr = x + (size_t)row * EMB_DIM;
    float s = 0.0f;
    #pragma unroll
    for (int i = 0; i < 4; ++i) {
        const int d = lane + i * 32;
        float e = er[d], xv = xr[d];
        quant[(size_t)row * EMB_DIM + d] = xv + (e - xv);  // straight-through forward value
        float t = e - xv;
        s += t * t;
    }
    #pragma unroll
    for (int off = 16; off >= 1; off >>= 1) s += __shfl_xor(s, off, 32);
    if (lane == 0) {
        wsum[wave] = s;
        atomicAdd(&counts[idx], 1u);
        enc[(size_t)row * NUM_EMB + idx] = 1.0f;
    }
    __syncthreads();
    if (threadIdx.x == 0) {
        float t = 0.0f;
        #pragma unroll
        for (int w = 0; w < 8; ++w) t += wsum[w];
        partials[blockIdx.x] = t;   // written unconditionally every call -> no stale state
    }
}

__global__ void __launch_bounds__(1024) k_scalars(const float* __restrict__ partials,
                                                  const unsigned* __restrict__ counts,
                                                  float* __restrict__ lossOut,
                                                  float* __restrict__ perpOut) {
    __shared__ float red[1024];
    const int t = threadIdx.x;
    float s = 0.0f;
    #pragma unroll
    for (int i = 0; i < 8; ++i) s += partials[t + i * 1024];
    red[t] = s;
    __syncthreads();
    for (int off = 512; off >= 1; off >>= 1) {
        if (t < off) red[t] += red[t + off];
        __syncthreads();
    }
    if (t == 0) lossOut[0] = red[0] / (float)((size_t)N_TOKENS * EMB_DIM);
    __syncthreads();
    float p = (float)counts[t] / (float)N_TOKENS;
    red[t] = p * logf(p + 1e-10f);
    __syncthreads();
    for (int off = 512; off >= 1; off >>= 1) {
        if (t < off) red[t] += red[t + off];
        __syncthreads();
    }
    if (t == 0) perpOut[0] = expf(-red[0]);
}

extern "C" void kernel_launch(void* const* d_in, const int* in_sizes, int n_in,
                              void* d_out, int out_size, void* d_ws, size_t ws_size,
                              hipStream_t stream) {
    const float* x   = (const float*)d_in[0];  // [65536,128]
    const float* emb = (const float*)d_in[1];  // [1024,128]

    float* out   = (float*)d_out;
    float* loss  = out;                                     // [0]
    float* quant = out + 1;                                 // [1 .. 1+65536*128)
    float* perp  = out + 1 + (size_t)N_TOKENS * EMB_DIM;    // scalar
    float* enc   = perp + 1;                                // [65536*1024]

    float*    esq      = (float*)d_ws;
    unsigned* counts   = (unsigned*)((float*)d_ws + 1024);
    float*    partials = (float*)d_ws + 2048;
    int*      idxArr   = (int*)((float*)d_ws + 2048 + 8192);
    unsigned* bhi      = (unsigned*)((float*)d_ws + 75776);
    unsigned* blo      = (unsigned*)((float*)d_ws + 141312);

    // zero 256MB encodings (b64 stores; enc byte offset is 8-aligned) + counts
    k_zero2<<<8192, 256, 0, stream>>>((float2*)enc, (long long)((size_t)N_TOKENS * NUM_EMB / 2));
    k_zero2<<<1, 256, 0, stream>>>((float2*)counts, 512);

    k_esumsq<<<NUM_EMB / 256, 256, 0, stream>>>(emb, esq);
    k_pack<<<(NUM_EMB * EMB_DIM / 2) / 256, 256, 0, stream>>>(emb, bhi, blo);
    k_argmin<<<N_TOKENS / 128, 256, 0, stream>>>(x, (const v16bf*)bhi, (const v16bf*)blo,
                                                 esq, idxArr);
    k_finalize<<<N_TOKENS / 8, 256, 0, stream>>>(x, emb, idxArr, quant, enc, counts, partials);
    k_scalars<<<1, 1024, 0, stream>>>(partials, counts, loss, perp);
}
